// PointFeatureNet_47150150975686
// MI455X (gfx1250) — matrix operand order repaired and checked by reading, not compiled, not run on hardware
//
#include <hip/hip_runtime.h>

// ---------------------------------------------------------------------------
// PointNet++-style SetConv x2 for MI455X (gfx1250, wave32, WMMA f16).
// Round 2: vectorized LDS fragment traffic.
//  - Weights pre-swizzled into WMMA B-fragment order -> v16h (2x ds_load_b128)
//  - A fragments loaded as two 16-byte runs (2x ds_load_b128)
// ---------------------------------------------------------------------------

typedef __attribute__((ext_vector_type(16))) _Float16 v16h;
typedef __attribute__((ext_vector_type(8)))  _Float16 v8h;
typedef __attribute__((ext_vector_type(8)))  float    v8f;

union AFrag { v16h v; v8h h[2]; _Float16 e[16]; };
union CFrag { v8f  v; float e[8]; };

constexpr int Bc  = 4;
constexpr int Nc  = 4096;
constexpr int S1c = 2048;   // N * 0.5
constexpr int S2c = 512;    // S1 * 0.25
constexpr int Kc  = 64;     // ball-query neighbors
constexpr int QPB = 8;      // queries per block in MLP kernels

// ---------------------------------------------------------------------------
// WMMA fragment helpers (16x16x32 f16 -> f32), layouts per CDNA5 ISA 7.12.2
// ---------------------------------------------------------------------------

// A-matrix 16x32 f16: lane L holds row M = L%16.
// lanes 0-15:  halves 0..7 -> K=0..7,   8..15 -> K=16..23
// lanes 16-31: halves 0..7 -> K=8..15,  8..15 -> K=24..31
// Both runs are 8 contiguous halves (16B, 16B-aligned for ldk in {32,64,96}).
__device__ __forceinline__ v16h load_a_frag(const _Float16* h, int row_base,
                                            int ldk, int ktile, int lane) {
  const _Float16* p = h + (row_base + (lane & 15)) * ldk + ktile * 32 +
                      ((lane < 16) ? 0 : 8);
  AFrag a;
  a.h[0] = *(const v8h*)(p);
  a.h[1] = *(const v8h*)(p + 16);
  return a.v;
}

// B fragments live in LDS pre-swizzled: fragment (kt,nt), lane L, half j at
// dst[((kt*NT+nt)*32 + L)*16 + j]  ->  one 32B contiguous load per lane.
__device__ __forceinline__ v16h load_b_swz(const _Float16* w, int frag,
                                           int lane) {
  return *(const v16h*)(w + ((size_t)(frag * 32 + lane) << 4));
}

// Swizzle global row-major weights [Kreal x ldn] (K padded to 32*KT) into
// B-fragment order.  b.e[j] of lane L of frag (kt,nt) holds
// W[kt*32 + (L<16?0:8) + j + (j<8?0:8)][nt*16 + L%16].
__device__ __forceinline__ void fill_b_swz(_Float16* dst,
                                           const float* __restrict__ w,
                                           int Kreal, int KT, int NT, int ldn,
                                           int tid, int nthreads) {
  const int total = KT * NT * 32 * 16;
  for (int idx = tid; idx < total; idx += nthreads) {
    const int j    = idx & 15;
    const int lane = (idx >> 4) & 31;
    const int frag = idx >> 9;
    const int nt   = frag % NT;
    const int kt   = frag / NT;
    const int k    = kt * 32 + ((lane < 16) ? 0 : 8) + j + ((j < 8) ? 0 : 8);
    const int n    = nt * 16 + (lane & 15);
    dst[idx] = (k < Kreal) ? (_Float16)w[k * ldn + n] : (_Float16)0.f;
  }
}

__device__ __forceinline__ v8f wmma16(v16h a, v16h b, v8f c) {
  // (neg_a, A, neg_b, B, c_mod, C, reuse_a, reuse_b)
  return __builtin_amdgcn_wmma_f32_16x16x32_f16(false, a, false, b,
                                                (short)0, c, false, false);
}

// D (16x16 f32): VGPR r holds row M = r + 8*(lane>=16), col N = lane%16.
// Fuses bias + ReLU, writes f16 activation row-major [rows][ldn].
__device__ __forceinline__ void store_d_relu(v8f cv, _Float16* h, int row_base,
                                             int ldn, int ntile,
                                             const float* bias, int lane) {
  CFrag c; c.v = cv;
  const int col = ntile * 16 + (lane & 15);
  const float bv = bias[col];
  const int rb = row_base + ((lane < 16) ? 0 : 8);
#pragma unroll
  for (int r = 0; r < 8; ++r) {
    float v = c.e[r] + bv;
    h[(rb + r) * ldn + col] = (_Float16)(v > 0.f ? v : 0.f);
  }
}

// Final layer: bias + ReLU + validity mask + max over this wave's 16 rows,
// reduced straight out of the C fragment (no LDS round trip for h3).
__device__ __forceinline__ void maxpool_frag(v8f cv, float* pw, int w,
                                             int ntile, int cout,
                                             const float* bias, int cnt,
                                             int lane) {
  CFrag c; c.v = cv;
  const int col = ntile * 16 + (lane & 15);
  const float bv = bias[col];
  const int rb = 16 * w + ((lane < 16) ? 0 : 8);
  float m = -__builtin_inff();
#pragma unroll
  for (int r = 0; r < 8; ++r) {
    float v = c.e[r] + bv;
    v = v > 0.f ? v : 0.f;
    m = (rb + r < cnt) ? fmaxf(m, v) : m;   // invalid rows -> -inf (ref semantics)
  }
  m = fmaxf(m, __shfl_xor(m, 16, 32));      // combine lane L with L+16
  if (lane < 16) pw[w * cout + col] = m;
}

// ---------------------------------------------------------------------------
// Kernel 0: split x[B][N][6] into pos[B][N][3], feat[B][N][3]
// ---------------------------------------------------------------------------
__global__ void split_kernel(const float* __restrict__ x,
                             float* __restrict__ pos0,
                             float* __restrict__ feat0) {
  int i = blockIdx.x * blockDim.x + threadIdx.x;
  if (i >= Bc * Nc) return;
  __builtin_prefetch(x + (i + 256) * 6, 0, 1);
#pragma unroll
  for (int c = 0; c < 3; ++c) {
    pos0[i * 3 + c]  = x[i * 6 + c];
    feat0[i * 3 + c] = x[i * 6 + 3 + c];
  }
}

// ---------------------------------------------------------------------------
// Kernel 1: farthest point sampling, one workgroup per cloud.
// ---------------------------------------------------------------------------
__global__ __launch_bounds__(1024) void fps_kernel(const float* __restrict__ pos,
                                                   int* __restrict__ idx,
                                                   int Npts, int S) {
  __shared__ float min_d[4096];
  __shared__ float rv[32];
  __shared__ int   ri[32];
  __shared__ float lp[3];
  __shared__ int   lastIdx;

  const int b = blockIdx.x;
  const float* P = pos + (size_t)b * Npts * 3;

  for (int i = threadIdx.x; i < Npts; i += blockDim.x)
    min_d[i] = __builtin_inff();
  if (threadIdx.x == 0) { idx[b * S] = 0; lastIdx = 0; }
  __syncthreads();

  for (int s = 1; s < S; ++s) {
    if (threadIdx.x == 0) {
      int l = lastIdx;
      lp[0] = P[l * 3 + 0]; lp[1] = P[l * 3 + 1]; lp[2] = P[l * 3 + 2];
    }
    __syncthreads();
    const float lx = lp[0], ly = lp[1], lz = lp[2];

    float best = -__builtin_inff();
    int   bi   = 0x7fffffff;
    for (int i = threadIdx.x; i < Npts; i += blockDim.x) {
      float dx = P[i * 3 + 0] - lx;
      float dy = P[i * 3 + 1] - ly;
      float dz = P[i * 3 + 2] - lz;
      float d  = dx * dx + dy * dy + dz * dz;
      float m  = fminf(min_d[i], d);
      min_d[i] = m;
      if (m > best) { best = m; bi = i; }    // ascending i: ties keep lowest
    }
    // wave32 argmax reduce (prefer lower index on tie -> matches jnp.argmax)
#pragma unroll
    for (int off = 16; off > 0; off >>= 1) {
      float ov = __shfl_down(best, off, 32);
      int   oi = __shfl_down(bi,   off, 32);
      if (ov > best || (ov == best && oi < bi)) { best = ov; bi = oi; }
    }
    const int wv = threadIdx.x >> 5;
    if ((threadIdx.x & 31) == 0) { rv[wv] = best; ri[wv] = bi; }
    __syncthreads();
    if (threadIdx.x == 0) {
      float bb = rv[0]; int bbi = ri[0];
      const int nw = (int)(blockDim.x >> 5);
      for (int ww = 1; ww < nw; ++ww)
        if (rv[ww] > bb || (rv[ww] == bb && ri[ww] < bbi)) { bb = rv[ww]; bbi = ri[ww]; }
      idx[b * S + s] = bbi;
      lastIdx = bbi;
    }
    __syncthreads();
  }
}

// ---------------------------------------------------------------------------
// Kernel 2: gather sampled positions
// ---------------------------------------------------------------------------
__global__ void gather_qpos(const float* __restrict__ pos,
                            const int* __restrict__ idx,
                            float* __restrict__ qpos, int Npts, int S,
                            int total) {
  int i = blockIdx.x * blockDim.x + threadIdx.x;
  if (i >= total) return;
  int b = i / S;
  int src = idx[i];
  const float* p = pos + ((size_t)b * Npts + src) * 3;
#pragma unroll
  for (int c = 0; c < 3; ++c) qpos[i * 3 + c] = p[c];
}

// ---------------------------------------------------------------------------
// Kernel 3: ball query — one wave32 per query; ballot/popc stream compaction
// keeps the first <=K lowest-index points within radius (== top_k(-arange)).
// ---------------------------------------------------------------------------
__global__ __launch_bounds__(256) void ball_query_kernel(
    const float* __restrict__ pos, const float* __restrict__ qpos,
    int* __restrict__ nidx, int* __restrict__ cnt_out,
    int Npts, int S, float r2) {
  const int gwave = (blockIdx.x * blockDim.x + threadIdx.x) >> 5;
  const int lane  = threadIdx.x & 31;
  const int b = gwave / S;
  const float qx = qpos[gwave * 3 + 0];
  const float qy = qpos[gwave * 3 + 1];
  const float qz = qpos[gwave * 3 + 2];
  const float* P = pos + (size_t)b * Npts * 3;

  int cnt = 0;
  for (int base = 0; base < Npts && cnt < Kc; base += 32) {
    __builtin_prefetch(P + (base + 32) * 3, 0, 1);
    const int i = base + lane;                       // Npts is multiple of 32
    float dx = P[i * 3 + 0] - qx;
    float dy = P[i * 3 + 1] - qy;
    float dz = P[i * 3 + 2] - qz;
    bool pred = (dx * dx + dy * dy + dz * dz) <= r2;
    unsigned m = (unsigned)(__ballot(pred) & 0xffffffffull);
    int pre = __popc(m & ((1u << lane) - 1u));
    if (pred && (cnt + pre) < Kc) nidx[gwave * Kc + cnt + pre] = i;
    cnt = min(Kc, cnt + __popc(m));
  }
  for (int j = cnt + lane; j < Kc; j += 32) nidx[gwave * Kc + j] = 0;
  if (lane == 0) cnt_out[gwave] = cnt;
}

// ---------------------------------------------------------------------------
// Kernel 4: SetConv layer 1 — group + MLP(6->32->32->64) + maxpool.
// 128 threads = 4 waves; wave w owns neighbor rows 16w..16w+15.
// ---------------------------------------------------------------------------
__global__ __launch_bounds__(128) void mlp1_kernel(
    const float* __restrict__ pos0, const float* __restrict__ feat0,
    const float* __restrict__ qpos, const int* __restrict__ nidx,
    const int* __restrict__ cnt,
    const float* __restrict__ w1a, const float* __restrict__ b1a,
    const float* __restrict__ w1b, const float* __restrict__ b1b,
    const float* __restrict__ w1c, const float* __restrict__ b1c,
    float* __restrict__ out /* [B*S1][64] */) {
  __shared__ __align__(32) _Float16 wa[1 * 2 * 512];   // KT=1, NT=2 (swizzled)
  __shared__ __align__(32) _Float16 wb[1 * 2 * 512];
  __shared__ __align__(32) _Float16 wc[1 * 4 * 512];
  __shared__ float ba[32], bb[32], bc[64];
  __shared__ __align__(32) _Float16 h0[64 * 32];
  __shared__ __align__(32) _Float16 h1[64 * 32];
  __shared__ __align__(32) _Float16 h2[64 * 32];
  __shared__ float pw[4 * 64];
  __shared__ float qp[3];
  __shared__ int scnt;

  const int tid = threadIdx.x, lane = tid & 31, w = tid >> 5;

  fill_b_swz(wa, w1a, /*Kreal=*/6,  /*KT=*/1, /*NT=*/2, /*ldn=*/32, tid, 128);
  fill_b_swz(wb, w1b, /*Kreal=*/32, /*KT=*/1, /*NT=*/2, /*ldn=*/32, tid, 128);
  fill_b_swz(wc, w1c, /*Kreal=*/32, /*KT=*/1, /*NT=*/4, /*ldn=*/64, tid, 128);
  if (tid < 32) { ba[tid] = b1a[tid]; bb[tid] = b1b[tid]; }
  if (tid < 64) bc[tid] = b1c[tid];
  __syncthreads();

  for (int qi = 0; qi < QPB; ++qi) {
    const int q = blockIdx.x * QPB + qi;     // [0, B*S1)
    const int b = q / S1c;
    if (tid == 0) scnt = cnt[q];
    if (tid < 3) qp[tid] = qpos[q * 3 + tid];
    __syncthreads();

    if (tid < 64) {
      const int ni = nidx[q * Kc + tid];
      const float* fp = feat0 + ((size_t)b * Nc + ni) * 3;
      const float* pp = pos0 + ((size_t)b * Nc + ni) * 3;
      _Float16* row = h0 + tid * 32;
      row[0] = (_Float16)fp[0];
      row[1] = (_Float16)fp[1];
      row[2] = (_Float16)fp[2];
      row[3] = (_Float16)(pp[0] - qp[0]);
      row[4] = (_Float16)(pp[1] - qp[1]);
      row[5] = (_Float16)(pp[2] - qp[2]);
#pragma unroll
      for (int c = 6; c < 32; ++c) row[c] = (_Float16)0.f;
    }
    __syncthreads();
    const int mycnt = scnt;

    // layer a: 6(pad32) -> 32
    {
      v16h a = load_a_frag(h0, 16 * w, 32, 0, lane);
#pragma unroll
      for (int nt = 0; nt < 2; ++nt) {
        v8f c = {};
        c = wmma16(a, load_b_swz(wa, nt, lane), c);
        store_d_relu(c, h1, 16 * w, 32, nt, ba, lane);
      }
    }
    __syncthreads();
    // layer b: 32 -> 32
    {
      v16h a = load_a_frag(h1, 16 * w, 32, 0, lane);
#pragma unroll
      for (int nt = 0; nt < 2; ++nt) {
        v8f c = {};
        c = wmma16(a, load_b_swz(wb, nt, lane), c);
        store_d_relu(c, h2, 16 * w, 32, nt, bb, lane);
      }
    }
    __syncthreads();
    // layer c: 32 -> 64, fused maxpool from C fragments
    {
      v16h a = load_a_frag(h2, 16 * w, 32, 0, lane);
#pragma unroll
      for (int nt = 0; nt < 4; ++nt) {
        v8f c = {};
        c = wmma16(a, load_b_swz(wc, nt, lane), c);
        maxpool_frag(c, pw, w, nt, 64, bc, mycnt, lane);
      }
    }
    __syncthreads();
    if (tid < 64) {
      float m = pw[tid];
#pragma unroll
      for (int ww = 1; ww < 4; ++ww) m = fmaxf(m, pw[ww * 64 + tid]);
      out[(size_t)q * 64 + tid] = m;
    }
    __syncthreads();
  }
}

// ---------------------------------------------------------------------------
// Kernel 5: SetConv layer 2 — group + MLP(67->64->64->128) + maxpool.
// ---------------------------------------------------------------------------
__global__ __launch_bounds__(128) void mlp2_kernel(
    const float* __restrict__ pos1 /* qpos1 [B][S1][3] */,
    const float* __restrict__ feat1 /* [B*S1][64] */,
    const float* __restrict__ qpos, const int* __restrict__ nidx,
    const int* __restrict__ cnt,
    const float* __restrict__ w2a, const float* __restrict__ b2a,
    const float* __restrict__ w2b, const float* __restrict__ b2b,
    const float* __restrict__ w2c, const float* __restrict__ b2c,
    float* __restrict__ out /* d_out feat region [B*S2][128] */) {
  __shared__ __align__(32) _Float16 wa[3 * 4 * 512];   // KT=3, NT=4 (12 KB)
  __shared__ __align__(32) _Float16 wb[2 * 4 * 512];   // KT=2, NT=4 (8 KB)
  __shared__ __align__(32) _Float16 wc[2 * 8 * 512];   // KT=2, NT=8 (16 KB)
  __shared__ float ba[64], bb[64], bc[128];
  __shared__ __align__(32) _Float16 buf0[64 * 96];     // h0, reused as h2
  __shared__ __align__(32) _Float16 h1[64 * 64];
  __shared__ float pw[4 * 128];
  __shared__ float qp[3];
  __shared__ int scnt;

  const int tid = threadIdx.x, lane = tid & 31, w = tid >> 5;

  fill_b_swz(wa, w2a, /*Kreal=*/67, /*KT=*/3, /*NT=*/4, /*ldn=*/64,  tid, 128);
  fill_b_swz(wb, w2b, /*Kreal=*/64, /*KT=*/2, /*NT=*/4, /*ldn=*/64,  tid, 128);
  fill_b_swz(wc, w2c, /*Kreal=*/64, /*KT=*/2, /*NT=*/8, /*ldn=*/128, tid, 128);
  if (tid < 64) { ba[tid] = b2a[tid]; bb[tid] = b2b[tid]; }
  bc[tid] = b2c[tid];                       // blockDim == 128
  __syncthreads();

  for (int qi = 0; qi < QPB; ++qi) {
    const int q = blockIdx.x * QPB + qi;    // [0, B*S2)
    const int b = q / S2c;
    if (tid == 0) scnt = cnt[q];
    if (tid < 3) qp[tid] = qpos[q * 3 + tid];
    __syncthreads();

    if (tid < 64) {
      const int ni = nidx[q * Kc + tid];
      const float* f = feat1 + ((size_t)b * S1c + ni) * 64;
      const float* p = pos1 + ((size_t)b * S1c + ni) * 3;
      _Float16* row = buf0 + tid * 96;
#pragma unroll
      for (int c = 0; c < 64; ++c) row[c] = (_Float16)f[c];
      row[64] = (_Float16)(p[0] - qp[0]);
      row[65] = (_Float16)(p[1] - qp[1]);
      row[66] = (_Float16)(p[2] - qp[2]);
#pragma unroll
      for (int c = 67; c < 96; ++c) row[c] = (_Float16)0.f;
    }
    __syncthreads();
    const int mycnt = scnt;

    // layer a: 67(pad96) -> 64   (3 K-tiles x 4 N-tiles)
    {
      v16h a0 = load_a_frag(buf0, 16 * w, 96, 0, lane);
      v16h a1 = load_a_frag(buf0, 16 * w, 96, 1, lane);
      v16h a2 = load_a_frag(buf0, 16 * w, 96, 2, lane);
#pragma unroll
      for (int nt = 0; nt < 4; ++nt) {
        v8f c = {};
        c = wmma16(a0, load_b_swz(wa, 0 * 4 + nt, lane), c);
        c = wmma16(a1, load_b_swz(wa, 1 * 4 + nt, lane), c);
        c = wmma16(a2, load_b_swz(wa, 2 * 4 + nt, lane), c);
        store_d_relu(c, h1, 16 * w, 64, nt, ba, lane);
      }
    }
    __syncthreads();
    // layer b: 64 -> 64   (h2 reuses h0's LDS)
    {
      _Float16* h2 = buf0;
      v16h a0 = load_a_frag(h1, 16 * w, 64, 0, lane);
      v16h a1 = load_a_frag(h1, 16 * w, 64, 1, lane);
#pragma unroll
      for (int nt = 0; nt < 4; ++nt) {
        v8f c = {};
        c = wmma16(a0, load_b_swz(wb, 0 * 4 + nt, lane), c);
        c = wmma16(a1, load_b_swz(wb, 1 * 4 + nt, lane), c);
        store_d_relu(c, h2, 16 * w, 64, nt, bb, lane);
      }
    }
    __syncthreads();
    // layer c: 64 -> 128, fused maxpool
    {
      const _Float16* h2 = buf0;
      v16h a0 = load_a_frag(h2, 16 * w, 64, 0, lane);
      v16h a1 = load_a_frag(h2, 16 * w, 64, 1, lane);
#pragma unroll
      for (int nt = 0; nt < 8; ++nt) {
        v8f c = {};
        c = wmma16(a0, load_b_swz(wc, 0 * 8 + nt, lane), c);
        c = wmma16(a1, load_b_swz(wc, 1 * 8 + nt, lane), c);
        maxpool_frag(c, pw, w, nt, 128, bc, mycnt, lane);
      }
    }
    __syncthreads();
    {
      float m = pw[tid];
#pragma unroll
      for (int ww = 1; ww < 4; ++ww) m = fmaxf(m, pw[ww * 128 + tid]);
      out[(size_t)q * 128 + tid] = m;
    }
    __syncthreads();
  }
}

// ---------------------------------------------------------------------------
// Kernel 6: write pos + batch tail of d_out
// ---------------------------------------------------------------------------
__global__ void finalize_kernel(const float* __restrict__ qpos2,
                                float* __restrict__ out) {
  const int FEAT = Bc * S2c * 128;   // 262144
  const int POS  = Bc * S2c * 3;     // 6144
  int i = blockIdx.x * blockDim.x + threadIdx.x;
  if (i < POS) out[FEAT + i] = qpos2[i];
  if (i < Bc * S2c) out[FEAT + POS + i] = (float)(i / S2c);
}

// ---------------------------------------------------------------------------
extern "C" void kernel_launch(void* const* d_in, const int* in_sizes, int n_in,
                              void* d_out, int out_size, void* d_ws,
                              size_t ws_size, hipStream_t stream) {
  const float* x   = (const float*)d_in[0];
  const float* w1a = (const float*)d_in[1];
  const float* b1a = (const float*)d_in[2];
  const float* w1b = (const float*)d_in[3];
  const float* b1b = (const float*)d_in[4];
  const float* w1c = (const float*)d_in[5];
  const float* b1c = (const float*)d_in[6];
  const float* w2a = (const float*)d_in[7];
  const float* b2a = (const float*)d_in[8];
  const float* w2b = (const float*)d_in[9];
  const float* b2b = (const float*)d_in[10];
  const float* w2c = (const float*)d_in[11];
  const float* b2c = (const float*)d_in[12];
  float* out = (float*)d_out;

  // workspace carve-up (~5.3 MB)
  float* wsf   = (float*)d_ws;
  float* pos0  = wsf;               wsf += Bc * Nc * 3;
  float* feat0 = wsf;               wsf += Bc * Nc * 3;
  int*   fps1  = (int*)wsf;         wsf += Bc * S1c;
  float* qpos1 = wsf;               wsf += Bc * S1c * 3;
  int*   nidx1 = (int*)wsf;         wsf += Bc * S1c * Kc;
  int*   cnt1  = (int*)wsf;         wsf += Bc * S1c;
  float* feat1 = wsf;               wsf += Bc * S1c * 64;
  int*   fps2  = (int*)wsf;         wsf += Bc * S2c;
  float* qpos2 = wsf;               wsf += Bc * S2c * 3;
  int*   nidx2 = (int*)wsf;         wsf += Bc * S2c * Kc;
  int*   cnt2  = (int*)wsf;         wsf += Bc * S2c;

  split_kernel<<<(Bc * Nc + 255) / 256, 256, 0, stream>>>(x, pos0, feat0);

  // ---- layer 1: r=0.5, S1=2048 ----
  fps_kernel<<<Bc, 1024, 0, stream>>>(pos0, fps1, Nc, S1c);
  gather_qpos<<<(Bc * S1c + 255) / 256, 256, 0, stream>>>(pos0, fps1, qpos1,
                                                          Nc, S1c, Bc * S1c);
  ball_query_kernel<<<(Bc * S1c) / 8, 256, 0, stream>>>(pos0, qpos1, nidx1,
                                                        cnt1, Nc, S1c, 0.25f);
  mlp1_kernel<<<(Bc * S1c) / QPB, 128, 0, stream>>>(
      pos0, feat0, qpos1, nidx1, cnt1, w1a, b1a, w1b, b1b, w1c, b1c, feat1);

  // ---- layer 2: r=1.0, S2=512 ----
  fps_kernel<<<Bc, 1024, 0, stream>>>(qpos1, fps2, S1c, S2c);
  gather_qpos<<<(Bc * S2c + 255) / 256, 256, 0, stream>>>(qpos1, fps2, qpos2,
                                                          S1c, S2c, Bc * S2c);
  ball_query_kernel<<<(Bc * S2c) / 8, 256, 0, stream>>>(qpos1, qpos2, nidx2,
                                                        cnt2, S1c, S2c, 1.0f);
  mlp2_kernel<<<(Bc * S2c) / QPB, 128, 0, stream>>>(
      qpos1, feat1, qpos2, nidx2, cnt2, w2a, b2a, w2b, b2b, w2c, b2c, out);

  finalize_kernel<<<(Bc * S2c * 3 + 255) / 256, 256, 0, stream>>>(qpos2, out);
}
